// ParallelMLPs_47571057770918
// MI455X (gfx1250) — compile-verified, matrix operand order checked by reading, not run on hardware
//
#include <hip/hip_runtime.h>

// ---- CDNA5 WMMA types ----
typedef __bf16        v16bf  __attribute__((ext_vector_type(16)));
typedef __bf16        bf16x2 __attribute__((ext_vector_type(2)));
typedef float         v8f    __attribute__((ext_vector_type(8)));
typedef unsigned int  uint4v __attribute__((ext_vector_type(4)));

union Frag {                 // A/B operand of v_wmma_f32_16x16x32_bf16 (8 VGPRs/lane)
  v16bf          v;
  __bf16         h[16];
  unsigned int   w[8];
  uint4v         q[2];
};

// f32 pair -> packed bf16x2 (one v_cvt_pk_bf16_f32 where available)
__device__ __forceinline__ unsigned pk2(float lo, float hi) {
  union { bf16x2 v; unsigned u; } c;
#if __has_builtin(__builtin_amdgcn_cvt_pk_bf16_f32)
  c.v = __builtin_amdgcn_cvt_pk_bf16_f32(lo, hi);
#else
  c.v[0] = (__bf16)lo; c.v[1] = (__bf16)hi;
#endif
  return c.u;
}

__device__ __forceinline__ float fast_sigmoid_mul(float v) {  // v * sigmoid(v)
#if __has_builtin(__builtin_amdgcn_exp2f) && __has_builtin(__builtin_amdgcn_rcpf)
  float e = __builtin_amdgcn_exp2f(v * -1.44269504088896f);   // exp(-v)
  return v * __builtin_amdgcn_rcpf(1.0f + e);
#else
  return v / (1.0f + __expf(-v));
#endif
}

#define Bn 128
#define Sn 4096
#define Dn 64
#define Hn 256
#define W1S 36    // uints per row of W1^T [Hn][72 bf16]  (72 = 64 + pad, keeps 16B align)
#define W2S 132   // uints per row of W2^T [Dn][264 bf16] (264 = 256 + pad)

__global__ __launch_bounds__(256) void
mlp_fused_kernel(const float* __restrict__ x, const float* __restrict__ W1,
                 const float* __restrict__ W2, float* __restrict__ out)
{
  // Phase-split LDS: wbuf holds W1^T (36864B) in phase A, W2^T (33792B) in phase B.
  __shared__ unsigned int wbuf[Hn * W1S];   // 36864 B
  __shared__ __bf16 scratch[8 * 256];       // 512 B per wave: 16x16 bf16, ROW-major

  const int tid  = threadIdx.x;
  const int wave = tid >> 5;
  const int lane = tid & 31;
  const int nn   = lane & 15;   // column (B/C) or row (A) within 16-tile
  const int hh   = lane >> 4;   // half-wave selector (K/M interleave)
  const int b    = blockIdx.y;
  const int m0   = blockIdx.x * 128 + wave * 16;

  // ---------------- stage W1^T (bf16) into LDS ----------------
  {
    const float* W1b = W1 + (size_t)b * Dn * Hn;   // [D][H], row-major
    const int n = tid;                              // 256 threads <-> 256 H-columns
    unsigned int* dst = wbuf + n * W1S;
    #pragma unroll 8
    for (int k = 0; k < Dn; k += 2) {               // coalesced global reads across tid
      float f0 = W1b[(size_t)k * Hn + n];
      float f1 = W1b[(size_t)(k + 1) * Hn + n];
      dst[k >> 1] = pk2(f0, f1);
    }
  }

  // ---------------- load x A-fragments (16x32, K over D=64) ----------------
  const float* xrow = x + ((size_t)b * Sn + m0 + nn) * Dn;
  Frag a[2];
  #pragma unroll
  for (int f = 0; f < 2; ++f) {
    float4 p0 = *(const float4*)(xrow + 32 * f + 8 * hh);
    float4 p1 = *(const float4*)(xrow + 32 * f + 8 * hh + 4);
    float4 p2 = *(const float4*)(xrow + 32 * f + 16 + 8 * hh);
    float4 p3 = *(const float4*)(xrow + 32 * f + 16 + 8 * hh + 4);
    a[f].w[0] = pk2(p0.x, p0.y);  a[f].w[1] = pk2(p0.z, p0.w);
    a[f].w[2] = pk2(p1.x, p1.y);  a[f].w[3] = pk2(p1.z, p1.w);
    a[f].w[4] = pk2(p2.x, p2.y);  a[f].w[5] = pk2(p2.z, p2.w);
    a[f].w[6] = pk2(p3.x, p3.y);  a[f].w[7] = pk2(p3.z, p3.w);
  }
  __syncthreads();

  // ---------------- GEMM1: h = silu(x * W1), kept as 8 A-fragments ----------------
  Frag hf[8];
  __bf16* scr = scratch + wave * 256;   // per-wave 16x16 bf16, row-major [row][col]
  #pragma unroll
  for (int t = 0; t < 16; ++t) {        // 16 n-tiles over H=256
    const unsigned int* wrow = wbuf + (16 * t + nn) * W1S;
    Frag b0, b1;
    b0.q[0] = *(const uint4v*)(wrow + 4 * hh);          // K  0..15 interleave
    b0.q[1] = *(const uint4v*)(wrow + 8 + 4 * hh);      // K 16..31
    b1.q[0] = *(const uint4v*)(wrow + 16 + 4 * hh);     // K 32..47
    b1.q[1] = *(const uint4v*)(wrow + 24 + 4 * hh);     // K 48..63
    v8f c = {0.f, 0.f, 0.f, 0.f, 0.f, 0.f, 0.f, 0.f};
    c = __builtin_amdgcn_wmma_f32_16x16x32_bf16(false, a[0].v, false, b0.v, (short)0, c, false, false);
    c = __builtin_amdgcn_wmma_f32_16x16x32_bf16(false, a[1].v, false, b1.v, (short)0, c, false, false);
    // SiLU in f32 -> bf16 scatter into row-major scratch (8x ds_store_b16, no packing ALU)
    #pragma unroll
    for (int r = 0; r < 8; ++r)
      scr[(8 * hh + r) * 16 + nn] = (__bf16)fast_sigmoid_mul(c[r]);
    // transpose read-back: lane = row m, 8 contiguous cols -> ONE ds_load_b128
    hf[t >> 1].q[t & 1] = *(const uint4v*)(scr + nn * 16 + 8 * hh);
  }
  __syncthreads();   // done with W1^T region

  // ---------------- stage W2^T (bf16) into the SAME LDS region ----------------
  {
    const float* W2b = W2 + (size_t)b * Hn * Dn;   // [H][D], row-major
    const int d  = tid & 63;
    const int c0 = (tid >> 6) * 64;                // H-chunk of 64 per thread group
    unsigned int* dst = wbuf + d * W2S;
    #pragma unroll 8
    for (int k = 0; k < 64; k += 2) {
      float f0 = W2b[(size_t)(c0 + k) * Dn + d];
      float f1 = W2b[(size_t)(c0 + k + 1) * Dn + d];
      dst[(c0 + k) >> 1] = pk2(f0, f1);
    }
  }
  __syncthreads();

  // ---------------- GEMM2: out = h * W2 + x (residual folded into accumulator) ----------------
  v8f acc[4];
  #pragma unroll
  for (int t2 = 0; t2 < 4; ++t2)
    #pragma unroll
    for (int r = 0; r < 8; ++r)
      acc[t2][r] = x[((size_t)b * Sn + m0 + 8 * hh + r) * Dn + 16 * t2 + nn];

  #pragma unroll
  for (int kk = 0; kk < 8; ++kk) {            // K over H=256
    #pragma unroll
    for (int t2 = 0; t2 < 4; ++t2) {          // D=64 output columns
      const unsigned int* wrow = wbuf + (16 * t2 + nn) * W2S + 16 * kk;
      Frag bb;
      bb.q[0] = *(const uint4v*)(wrow + 4 * hh);
      bb.q[1] = *(const uint4v*)(wrow + 8 + 4 * hh);
      acc[t2] = __builtin_amdgcn_wmma_f32_16x16x32_bf16(false, hf[kk].v, false, bb.v,
                                                        (short)0, acc[t2], false, false);
    }
  }

  float* orow = out + ((size_t)b * Sn + m0) * Dn;
  #pragma unroll
  for (int t2 = 0; t2 < 4; ++t2)
    #pragma unroll
    for (int r = 0; r < 8; ++r)
      orow[(size_t)(8 * hh + r) * Dn + 16 * t2 + nn] = acc[t2][r];
}

extern "C" void kernel_launch(void* const* d_in, const int* in_sizes, int n_in,
                              void* d_out, int out_size, void* d_ws, size_t ws_size,
                              hipStream_t stream) {
  const float* x  = (const float*)d_in[0];
  const float* W1 = (const float*)d_in[1];
  const float* W2 = (const float*)d_in[2];
  float* out = (float*)d_out;
  dim3 grid(Sn / 128, Bn);   // 32 x 128 blocks, 8 waves each
  mlp_fused_kernel<<<grid, 256, 0, stream>>>(x, W1, W2, out);
}
